// MultiHeadSelfAttention_51797305590119
// MI455X (gfx1250) — matrix operand order repaired and checked by reading, not compiled, not run on hardware
//
#include <hip/hip_runtime.h>
#include <hip/hip_bf16.h>
#include <math.h>

// ---------------------------------------------------------------------------
// MHSA for MI455X (gfx1250, wave32).
//  - all matmuls on v_wmma_f32_16x16x32_f16 (f32 accumulate)
//  - flash-attention fusion (scores never hit HBM)
//  - K/V tiles staged to LDS with GLOBAL_LOAD_ASYNC_TO_LDS_B128, double
//    buffered on ASYNCcnt, shared by all 8 waves of the workgroup
// ---------------------------------------------------------------------------

typedef _Float16 f16;
typedef __attribute__((ext_vector_type(16))) _Float16 v16h;
typedef __attribute__((ext_vector_type(8)))  _Float16 v8h;
typedef __attribute__((ext_vector_type(8)))  float    v8f;
typedef __attribute__((ext_vector_type(4)))  float    v4f;

#define BB  4
#define CC  512
#define NH  8
#define HD  64
#define SS  2304      // 48*48
#define OC3 1536

union AF { v16h v; v8h h[2]; };

__device__ __forceinline__ v8f wmma_f16(v16h a, v16h b, v8f c) {
  return __builtin_amdgcn_wmma_f32_16x16x32_f16(
      false, a, false, b, (short)0, c, false, false);
}

// Async copy 16 bytes global -> LDS (per-lane addresses), tracked by ASYNCcnt.
// Generic LDS pointers carry the byte offset in their low 32 bits (the shared
// aperture lives entirely in addr[63:32] per the flat-addressing rules).
__device__ __forceinline__ void async_copy16(void* lds_dst, const void* gsrc) {
  asm volatile("global_load_async_to_lds_b128 %0, %1, off"
               :: "v"((unsigned)(unsigned long long)lds_dst), "v"(gsrc)
               : "memory");
}

// ---------------------------------------------------------------------------
// Kernel 0a: x (B,C,S) f32 -> xt (B,S,C) f16, tiled transpose through LDS
// ---------------------------------------------------------------------------
__global__ __launch_bounds__(256)
void k_transpose_x(const float* __restrict__ x, f16* __restrict__ xt) {
  __shared__ float tile[32][33];
  const int b  = blockIdx.z;
  const int s0 = blockIdx.x * 32;
  const int c0 = blockIdx.y * 32;
  const int tx = threadIdx.x;   // 0..31
  const int ty = threadIdx.y;   // 0..7
  const float* xp = x + ((size_t)b * CC + c0) * SS + s0;
  #pragma unroll
  for (int j = 0; j < 32; j += 8)
    tile[ty + j][tx] = xp[(size_t)(ty + j) * SS + tx];
  __syncthreads();
  f16* xo = xt + ((size_t)b * SS + s0) * CC + c0;
  #pragma unroll
  for (int j = 0; j < 32; j += 8)
    xo[(size_t)(ty + j) * CC + tx] = (f16)tile[tx][ty + j];
}

__global__ void k_convert(const float* __restrict__ in, f16* __restrict__ out, int n) {
  int i = blockIdx.x * blockDim.x + threadIdx.x;
  if (i < n) out[i] = (f16)in[i];
}

// ---------------------------------------------------------------------------
// GEMM core: one wave -> 16x64 strip (4 accumulators), K=512 fully unrolled.
// One A fragment is amortized over 4 B fragments (halves fragment traffic).
// ---------------------------------------------------------------------------
__device__ __forceinline__ void gemm4_k512(const f16* __restrict__ arow,
                                           const f16* __restrict__ const* brow,
                                           int klo, int klo2, v8f acc[4]) {
  #pragma unroll
  for (int kk = 0; kk < 512; kk += 32) {
    AF a;
    a.h[0] = *(const v8h*)(arow + kk + klo);
    a.h[1] = *(const v8h*)(arow + kk + 16 + klo);
    #pragma unroll
    for (int j = 0; j < 4; ++j) {
      AF bfr;
      bfr.h[0] = *(const v8h*)(brow[j] + kk + klo);
      bfr.h[1] = *(const v8h*)(brow[j] + kk + 16 + klo);
      acc[j] = wmma_f16(a.v, bfr.v, acc[j]);
    }
  }
}

// ---------------------------------------------------------------------------
// Kernel 1: QKV projection.  Q,K stored [b,h,s,d] f16; V stored [b,h,d,s] f16.
// ---------------------------------------------------------------------------
__global__ __launch_bounds__(256)
void k_qkv(const f16* __restrict__ W, const float* __restrict__ bias,
           const f16* __restrict__ xt,
           f16* __restrict__ qws, f16* __restrict__ kws, f16* __restrict__ vws) {
  const int widx = threadIdx.x >> 5;
  const int lane = threadIdx.x & 31;
  const int lmod = lane & 15;
  const int moff = (lane < 16) ? 0 : 8;

  int tile = blockIdx.x * 8 + widx;            // 4 * 96 * 36 strips
  int b    = tile / (96 * 36);
  int rem  = tile - b * (96 * 36);
  int mt   = rem / 36;
  int ng   = rem - mt * 36;

  const f16* arow = W + (size_t)(mt * 16 + lmod) * CC;
  const f16* brow[4];
  #pragma unroll
  for (int j = 0; j < 4; ++j)
    brow[j] = xt + ((size_t)b * SS + ng * 64 + j * 16 + lmod) * CC;

  v8f acc[4] = {};
  gemm4_k512(arow, brow, moff, moff * 2, acc);

  v4f b0 = *(const v4f*)(bias + mt * 16 + moff);
  v4f b1 = *(const v4f*)(bias + mt * 16 + moff + 4);
  #pragma unroll
  for (int j = 0; j < 4; ++j)
    #pragma unroll
    for (int r = 0; r < 4; ++r) { acc[j][r] += b0[r]; acc[j][r + 4] += b1[r]; }

  const int obase = mt * 16;
  const int qkvi  = obase >> 9;         // 0:q 1:k 2:v
  const int rem2  = obase & 511;
  const int h     = rem2 >> 6;
  const int dbase = rem2 & 63;
  const size_t bh = (size_t)b * NH + h;

  #pragma unroll
  for (int j = 0; j < 4; ++j) {
    const int s = ng * 64 + j * 16 + lmod;
    if (qkvi == 2) {
      #pragma unroll
      for (int r = 0; r < 8; ++r)
        vws[(bh * HD + dbase + moff + r) * SS + s] = (f16)acc[j][r];
    } else {
      f16* dst = (qkvi == 0 ? qws : kws) + (bh * SS + s) * HD + dbase + moff;
      v8h ov;
      #pragma unroll
      for (int r = 0; r < 8; ++r) ov[r] = (f16)acc[j][r];
      *(v8h*)dst = ov;
    }
  }
}

// ---------------------------------------------------------------------------
// Kernel 2: flash attention.  Workgroup = 8 waves on one (b,h), 256 queries.
// 64-key K/V tiles staged to LDS via async-to-LDS (double buffered); each
// wave owns 32 queries (2 q-tiles), online softmax, P through per-wave LDS.
// ---------------------------------------------------------------------------
__global__ __launch_bounds__(256)
void k_attn(const f16* __restrict__ q, const f16* __restrict__ k,
            const f16* __restrict__ v, f16* __restrict__ attn) {
  __shared__ __align__(16) f16 kt[2][64][64];       // [buf][t][d]   16KB
  __shared__ __align__(16) f16 vt[2][64][64];       // [buf][d][t]   16KB
  __shared__ __align__(16) f16 pls[8][16][64];      // per-wave P    16KB

  const int tid  = threadIdx.x;
  const int widx = tid >> 5;
  const int lane = tid & 31;
  const int lmod = lane & 15;
  const int moff = (lane < 16) ? 0 : 8;
  const int klo2 = moff * 2;

  const int bh = blockIdx.x / 9;        // 0..31
  const int qg = blockIdx.x - bh * 9;
  const int b  = bh >> 3, h = bh & 7;
  const int q0 = qg * 256 + widx * 32;

  const f16* kbase = k + (size_t)bh * SS * HD;
  const f16* vbase = v + (size_t)bh * HD * SS;

  // Q A-fragments: 2 q-tiles x 2 d-chunks
  AF aq[2][2];
  #pragma unroll
  for (int qi = 0; qi < 2; ++qi) {
    const f16* qrow = q + ((size_t)bh * SS + q0 + qi * 16 + lmod) * HD;
    aq[qi][0].h[0] = *(const v8h*)(qrow + moff);
    aq[qi][0].h[1] = *(const v8h*)(qrow + 16 + moff);
    aq[qi][1].h[0] = *(const v8h*)(qrow + 32 + moff);
    aq[qi][1].h[1] = *(const v8h*)(qrow + 48 + moff);
  }

  float mrow[2][8], lrow[2][8];
  v8f o[2][4] = {};
  #pragma unroll
  for (int qi = 0; qi < 2; ++qi)
    #pragma unroll
    for (int r = 0; r < 8; ++r) { mrow[qi][r] = -1e30f; lrow[qi][r] = 0.0f; }

  const float scale = 0.125f;                 // HD^-0.5
  const float L2E   = 1.44269504088896f;
  f16* pl = &pls[widx][0][0];

  // stage one 64-key K tile + V tile into LDS buffer nb (4 async ops/wave)
  auto stage = [&](int tn, int nb) {
    #pragma unroll
    for (int u = 0; u < 2; ++u) {
      int idx = tid + u * 256;            // 0..511
      int row = idx >> 3;                 // 0..63
      int col = (idx & 7) * 8;            // f16 column, 16B chunks
      async_copy16(&kt[nb][row][col], kbase + (size_t)(tn + row) * HD + col);
      async_copy16(&vt[nb][row][col], vbase + (size_t)row * SS + tn + col);
    }
  };

  stage(0, 0);
  int cb = 0;

  for (int t0 = 0; t0 < SS; t0 += 64) {
    if (t0 + 64 < SS) {
      stage(t0 + 64, cb ^ 1);
      asm volatile("s_wait_asynccnt 0x4" ::: "memory");   // current tile done
    } else {
      asm volatile("s_wait_asynccnt 0x0" ::: "memory");
    }
    __syncthreads();

    const f16* kbuf = &kt[cb][0][0];
    const f16* vbuf = &vt[cb][0][0];

    // scores: 16 WMMAs (K fragments shared across the 2 q-tiles)
    v8f sc[2][4];
    #pragma unroll
    for (int n4 = 0; n4 < 4; ++n4) {
      const f16* kr = kbuf + (size_t)(n4 * 16 + lmod) * 64;
      AF bk0, bk1;
      bk0.h[0] = *(const v8h*)(kr + moff);      bk0.h[1] = *(const v8h*)(kr + 16 + moff);
      bk1.h[0] = *(const v8h*)(kr + 32 + moff); bk1.h[1] = *(const v8h*)(kr + 48 + moff);
      #pragma unroll
      for (int qi = 0; qi < 2; ++qi) {
        v8f t = {};
        t = wmma_f16(aq[qi][0].v, bk0.v, t);
        t = wmma_f16(aq[qi][1].v, bk1.v, t);
        sc[qi][n4] = t;
      }
    }

    // online softmax + P*V per q-tile
    #pragma unroll
    for (int qi = 0; qi < 2; ++qi) {
      float al[8];
      #pragma unroll
      for (int r = 0; r < 8; ++r) {
        float rm = -1e30f;
        #pragma unroll
        for (int n4 = 0; n4 < 4; ++n4) {
          sc[qi][n4][r] *= scale;
          rm = fmaxf(rm, sc[qi][n4][r]);
        }
        rm = fmaxf(rm, __shfl_xor(rm, 1, 16));
        rm = fmaxf(rm, __shfl_xor(rm, 2, 16));
        rm = fmaxf(rm, __shfl_xor(rm, 4, 16));
        rm = fmaxf(rm, __shfl_xor(rm, 8, 16));
        float mn = fmaxf(mrow[qi][r], rm);
        al[r] = __builtin_amdgcn_exp2f((mrow[qi][r] - mn) * L2E);
        mrow[qi][r] = mn;
        float rs = 0.0f;
        #pragma unroll
        for (int n4 = 0; n4 < 4; ++n4) {
          float p = __builtin_amdgcn_exp2f((sc[qi][n4][r] - mn) * L2E);
          sc[qi][n4][r] = p;
          rs += p;
        }
        rs += __shfl_xor(rs, 1, 16);
        rs += __shfl_xor(rs, 2, 16);
        rs += __shfl_xor(rs, 4, 16);
        rs += __shfl_xor(rs, 8, 16);
        lrow[qi][r] = lrow[qi][r] * al[r] + rs;
      }
      #pragma unroll
      for (int i = 0; i < 4; ++i)
        #pragma unroll
        for (int r = 0; r < 8; ++r) o[qi][i][r] *= al[r];

      // P: C-fragment layout -> LDS (16x64) -> A-fragment layout
      #pragma unroll
      for (int r = 0; r < 8; ++r)
        #pragma unroll
        for (int n4 = 0; n4 < 4; ++n4)
          pl[(moff + r) * 64 + n4 * 16 + lmod] = (f16)sc[qi][n4][r];
      asm volatile("s_wait_dscnt 0x0" ::: "memory");    // wave-local RAW fence

      AF ap0, ap1;
      ap0.h[0] = *(const v8h*)(pl + lmod * 64 + moff);
      ap0.h[1] = *(const v8h*)(pl + lmod * 64 + 16 + moff);
      ap1.h[0] = *(const v8h*)(pl + lmod * 64 + 32 + moff);
      ap1.h[1] = *(const v8h*)(pl + lmod * 64 + 48 + moff);

      #pragma unroll
      for (int j = 0; j < 2; ++j) {
        const AF& ap = j ? ap1 : ap0;
        #pragma unroll
        for (int i = 0; i < 4; ++i) {
          const f16* vr = vbuf + (size_t)(i * 16 + lmod) * 64 + j * 32 + moff;
          AF bv;
          bv.h[0] = *(const v8h*)(vr);
          bv.h[1] = *(const v8h*)(vr + 16);
          o[qi][i] = wmma_f16(ap.v, bv.v, o[qi][i]);
        }
      }
    }

    __syncthreads();       // all waves done with buffer cb before it refills
    cb ^= 1;
  }

  #pragma unroll
  for (int qi = 0; qi < 2; ++qi) {
    float inv[8];
    #pragma unroll
    for (int r = 0; r < 8; ++r) inv[r] = 1.0f / lrow[qi][r];
    f16* aout = attn + ((size_t)b * SS + q0 + qi * 16 + moff) * CC + h * HD + lmod;
    #pragma unroll
    for (int i = 0; i < 4; ++i)
      #pragma unroll
      for (int r = 0; r < 8; ++r)
        aout[(size_t)r * CC + i * 16] = (f16)(o[qi][i][r] * inv[r]);
  }
}

// ---------------------------------------------------------------------------
// Kernel 3: output projection -> f32 output (+bias)
// ---------------------------------------------------------------------------
__global__ __launch_bounds__(256)
void k_out(const f16* __restrict__ W, const float* __restrict__ bias,
           const f16* __restrict__ attn, float* __restrict__ out) {
  const int widx = threadIdx.x >> 5;
  const int lane = threadIdx.x & 31;
  const int lmod = lane & 15;
  const int moff = (lane < 16) ? 0 : 8;

  int tile = blockIdx.x * 8 + widx;           // 4 * 32 * 36 strips
  int b    = tile / (32 * 36);
  int rem  = tile - b * (32 * 36);
  int mt   = rem / 36;
  int ng   = rem - mt * 36;

  const f16* arow = W + (size_t)(mt * 16 + lmod) * CC;
  const f16* brow[4];
  #pragma unroll
  for (int j = 0; j < 4; ++j)
    brow[j] = attn + ((size_t)b * SS + ng * 64 + j * 16 + lmod) * CC;

  v8f acc[4] = {};
  gemm4_k512(arow, brow, moff, moff * 2, acc);

  v4f b0 = *(const v4f*)(bias + mt * 16 + moff);
  v4f b1 = *(const v4f*)(bias + mt * 16 + moff + 4);
  #pragma unroll
  for (int j = 0; j < 4; ++j) {
    const int s = ng * 64 + j * 16 + lmod;
    float* dst = out + ((size_t)b * CC + mt * 16 + moff) * SS + s;
    #pragma unroll
    for (int r = 0; r < 8; ++r)
      dst[(size_t)r * SS] = acc[j][r] + ((r < 4) ? b0[r] : b1[r - 4]);
  }
}

// ---------------------------------------------------------------------------
// Host launch
// ---------------------------------------------------------------------------
extern "C" void kernel_launch(void* const* d_in, const int* in_sizes, int n_in,
                              void* d_out, int out_size, void* d_ws, size_t ws_size,
                              hipStream_t stream) {
  const float* x    = (const float*)d_in[0];
  const float* Wqkv = (const float*)d_in[1];
  const float* bqkv = (const float*)d_in[2];
  const float* Wout = (const float*)d_in[3];
  const float* bout = (const float*)d_in[4];
  float* out = (float*)d_out;

  char* ws = (char*)d_ws;
  const size_t XT_B   = (size_t)BB * SS * CC * 2;
  const size_t WQ_B   = (size_t)OC3 * CC * 2;
  const size_t WO_B   = (size_t)CC * CC * 2;
  const size_t HEAD_B = (size_t)BB * NH * SS * HD * 2;

  f16* xt  = (f16*)(ws);
  f16* wq  = (f16*)(ws + XT_B);
  f16* wo  = (f16*)(ws + XT_B + WQ_B);
  f16* qws = (f16*)(ws + XT_B + WQ_B + WO_B);
  f16* kws = (f16*)((char*)qws + HEAD_B);
  f16* vws = (f16*)((char*)kws + HEAD_B);
  f16* aws = (f16*)((char*)vws + HEAD_B);   // ~47 MB total workspace

  k_transpose_x<<<dim3(SS / 32, CC / 32, BB), dim3(32, 8, 1), 0, stream>>>(x, xt);
  k_convert<<<(OC3 * CC + 255) / 256, 256, 0, stream>>>(Wqkv, wq, OC3 * CC);
  k_convert<<<(CC * CC + 255) / 256, 256, 0, stream>>>(Wout, wo, CC * CC);

  k_qkv<<<(BB * 96 * 36) / 8, 256, 0, stream>>>(wq, bqkv, xt, qws, kws, vws);
  k_attn<<<32 * 9, 256, 0, stream>>>(qws, kws, vws, aws);
  k_out<<<(BB * 32 * 36) / 8, 256, 0, stream>>>(wo, bout, aws, out);
}
